// HierCDF_24111946400051
// MI455X (gfx1250) — matrix operand order repaired and checked by reading, not compiled, not run on hardware
//
#include <hip/hip_runtime.h>
#include <hip/hip_bf16.h>
#include <stdint.h>

typedef __attribute__((ext_vector_type(16))) _Float16 v16h;
typedef __attribute__((ext_vector_type(8)))  _Float16 v8h;
typedef __attribute__((ext_vector_type(8)))  float    v8f;

#define BATCH  8192
#define NKNOW  128
#define NEDGE  252
#define HIDDEN 512
#define H2     256

__device__ __forceinline__ float sigmoidf_(float x) {
    return 1.0f / (1.0f + __expf(-x));
}

// CDNA5 async copy: global -> LDS, tracked by ASYNCcnt (ISA 15.18.3 op 98).
__device__ __forceinline__ void async_b128(uint32_t lds_off, const _Float16* g) {
    asm volatile("global_load_async_to_lds_b128 %0, %1, off"
                 :: "v"(lds_off), "v"(g) : "memory");
}
__device__ __forceinline__ void wait_async0() {
    asm volatile("s_wait_asynccnt 0" ::: "memory");
}

// ---------------------------------------------------------------------------
// Kernel 0: f32 -> f16 weight conversion
// ---------------------------------------------------------------------------
__global__ void convert_f16_kernel(const float* __restrict__ src,
                                   _Float16* __restrict__ dst, int n) {
    int i = blockIdx.x * blockDim.x + threadIdx.x;
    if (i < n) dst[i] = (_Float16)src[i];
}

// ---------------------------------------------------------------------------
// Kernel 1: gathers + sigmoid + DAG posterior scan + Q-mask, emit f16 operands
// ---------------------------------------------------------------------------
__global__ void prep_kernel(const int* __restrict__ uid, const int* __restrict__ iid,
                            const float* __restrict__ know,
                            const float* __restrict__ priori,
                            const float* __restrict__ condi_p,
                            const float* __restrict__ condi_n,
                            const float* __restrict__ item_diff,
                            const float* __restrict__ item_disc,
                            _Float16* __restrict__ Xu, _Float16* __restrict__ Xi,
                            float* __restrict__ disc) {
    int b = blockIdx.x * blockDim.x + threadIdx.x;
    if (b >= BATCH) return;
    int u  = uid[b];
    int it = iid[b];
    disc[b] = sigmoidf_(item_disc[it]);

    const float* pr  = priori    + (size_t)u  * NKNOW;
    const float* cpb = condi_p   + (size_t)u  * NEDGE;
    const float* cnb = condi_n   + (size_t)u  * NEDGE;
    const float* kw  = know      + (size_t)b  * NKNOW;
    const float* idf = item_diff + (size_t)it * NKNOW;
    _Float16* xu = Xu + (size_t)b * NKNOW;
    _Float16* xi = Xi + (size_t)b * NKNOW;

    float pm2 = sigmoidf_(pr[0]);
    float pm1 = sigmoidf_(pr[1]);
    xu[0] = (_Float16)(pm2 * kw[0]);
    xu[1] = (_Float16)(pm1 * kw[1]);
    xi[0] = (_Float16)(sigmoidf_(idf[0]) * kw[0]);
    xi[1] = (_Float16)(sigmoidf_(idf[1]) * kw[1]);

    for (int k = 2; k < NKNOW; ++k) {
        int e = 2 * (k - 2);
        float cp0 = __fsqrt_rn(sigmoidf_(cpb[e]));
        float cp1 = __fsqrt_rn(sigmoidf_(cpb[e + 1]));
        float cn0 = __fsqrt_rn(sigmoidf_(cnb[e]));
        float cn1 = __fsqrt_rn(sigmoidf_(cnb[e + 1]));
        float m0 = cp0 * pm2 + cn0 * (1.0f - pm2);
        float m1 = cp1 * pm1 + cn1 * (1.0f - pm1);
        float pk = m0 * m1;
        float kk = kw[k];
        xu[k] = (_Float16)(pk * kk);
        xi[k] = (_Float16)(sigmoidf_(idf[k]) * kk);
        pm2 = pm1;
        pm1 = pk;
    }
}

// ---------------------------------------------------------------------------
// WMMA fragment loaders (CDNA5 wave32 16x16x32 f16 layouts)
// ---------------------------------------------------------------------------
// A (16x32 MxK): lanes 0-15 M=lane | 16-31 M=lane-16; K halves {0|8}+v,{16|24}+v
__device__ __forceinline__ v16h load_a_frag(const _Float16* __restrict__ A,
                                            int m0, int k0, int ld, int lane) {
    int m    = m0 + (lane & 15);
    int koff = (lane & 16) ? 8 : 0;
    const _Float16* p = A + (size_t)m * ld + k0 + koff;
    union { v16h v; struct { v8h lo; v8h hi; } s; } u;
    u.s.lo = *(const v8h*)(p);
    u.s.hi = *(const v8h*)(p + 16);
    return u.v;
}

// B (32x16 KxN) from an LDS tile with row stride ld (halves): lane holds
// column n = lane&15; K halves {0|8}+v,{16|24}+v (same as A) -> two b128.
__device__ __forceinline__ v16h lds_b_frag(const _Float16* W,
                                           int n0, int k0, int ld, int lane) {
    const _Float16* p = W + (size_t)(n0 + (lane & 15)) * ld + k0 + ((lane & 16) >> 1);
    union { v16h v; struct { v8h lo; v8h hi; } s; } u;
    u.s.lo = *(const v8h*)(p);
    u.s.hi = *(const v8h*)(p + 16);
    return u.v;
}

#define LD1 (NKNOW + 8)    // 136 halves: padded LDS row stride (gemm1)
#define LD2 (HIDDEN + 8)   // 520 halves: padded LDS row stride (gemm2)

// ---------------------------------------------------------------------------
// Kernel 2: fused layer 1. Block = 4 waves sharing one 64-col N-block of
// Wu/Wi staged in LDS by async copy; each wave owns a 16-row M-tile.
// ---------------------------------------------------------------------------
__global__ void __launch_bounds__(128)
gemm1_kernel(const _Float16* __restrict__ Xu, const _Float16* __restrict__ Xi,
             const _Float16* __restrict__ Wu, const _Float16* __restrict__ Wi,
             const float* __restrict__ b_user, const float* __restrict__ b_item,
             const float* __restrict__ disc, _Float16* __restrict__ X1) {
    __shared__ _Float16 lWu[64 * LD1];
    __shared__ _Float16 lWi[64 * LD1];

    int tid  = threadIdx.x;
    int lane = tid & 31;
    int wave = tid >> 5;
    int m0 = (blockIdx.x >> 3) * 64 + wave * 16;  // 128 M-blocks of 64 rows
    int h0 = (blockIdx.x & 7) * 64;               // 8 N-blocks of 64 cols

    // Stage W tiles (64 x 128 f16 each) into LDS with async b128 copies.
    uint32_t lu = (uint32_t)(size_t)lWu;   // low 32 bits == LDS byte offset
    uint32_t li = (uint32_t)(size_t)lWi;
    for (int c = tid; c < 64 * 16; c += 128) {   // 16 b128 chunks per row
        int row = c >> 4, ch = c & 15;
        uint32_t lo = (uint32_t)(row * LD1 * 2 + ch * 16);
        async_b128(lu + lo, Wu + (size_t)(h0 + row) * NKNOW + ch * 8);
        async_b128(li + lo, Wi + (size_t)(h0 + row) * NKNOW + ch * 8);
    }
    wait_async0();
    __syncthreads();

    v8f zero = {0.f, 0.f, 0.f, 0.f, 0.f, 0.f, 0.f, 0.f};
    v8f au[4] = {zero, zero, zero, zero};
    v8f ai[4] = {zero, zero, zero, zero};

    // K loop: A double-buffered from global, B fragments from LDS.
    v16h aU = load_a_frag(Xu, m0, 0, NKNOW, lane);
    v16h aI = load_a_frag(Xi, m0, 0, NKNOW, lane);
#pragma unroll
    for (int kk = 0; kk < NKNOW; kk += 32) {
        v16h aUn = aU, aIn = aI;
        if (kk + 32 < NKNOW) {
            aUn = load_a_frag(Xu, m0, kk + 32, NKNOW, lane);
            aIn = load_a_frag(Xi, m0, kk + 32, NKNOW, lane);
        }
#pragma unroll
        for (int j = 0; j < 4; ++j) {
            v16h bU = lds_b_frag(lWu, 16 * j, kk, LD1, lane);
            au[j] = __builtin_amdgcn_wmma_f32_16x16x32_f16(
                false, aU, false, bU, (short)0, au[j], false, false);
            v16h bI = lds_b_frag(lWi, 16 * j, kk, LD1, lane);
            ai[j] = __builtin_amdgcn_wmma_f32_16x16x32_f16(
                false, aI, false, bI, (short)0, ai[j], false, false);
        }
        aU = aUn;
        aI = aIn;
    }

    int n  = lane & 15;
    int mo = (lane & 16) ? 8 : 0;
    float dv[8];
#pragma unroll
    for (int v = 0; v < 8; ++v) dv[v] = disc[m0 + mo + v];
#pragma unroll
    for (int j = 0; j < 4; ++j) {
        int col = h0 + 16 * j + n;
        float bu = b_user[col];
        float bi = b_item[col];
#pragma unroll
        for (int v = 0; v < 8; ++v) {
            int row = m0 + mo + v;
            float uf  = tanhf(au[j][v] + bu);
            float itf = sigmoidf_(ai[j][v] + bi);
            X1[(size_t)row * HIDDEN + col] = (_Float16)((uf - itf) * dv[v]);
        }
    }
}

// ---------------------------------------------------------------------------
// Kernel 3: layer 2 — X2 = sigmoid(X1 @ W1^T + b1). W1 N-block (64x512 f16 =
// 64KB) staged once per block in LDS.
// ---------------------------------------------------------------------------
__global__ void __launch_bounds__(128)
gemm2_kernel(const _Float16* __restrict__ X1, const _Float16* __restrict__ W1h,
             const float* __restrict__ b1, _Float16* __restrict__ X2) {
    __shared__ _Float16 lW1[64 * LD2];

    int tid  = threadIdx.x;
    int lane = tid & 31;
    int wave = tid >> 5;
    int m0 = (blockIdx.x >> 2) * 64 + wave * 16;  // 128 M-blocks
    int h0 = (blockIdx.x & 3) * 64;               // 4 N-blocks

    uint32_t lw = (uint32_t)(size_t)lW1;
    for (int c = tid; c < 64 * 64; c += 128) {    // 64 b128 chunks per row
        int row = c >> 6, ch = c & 63;
        async_b128(lw + (uint32_t)(row * LD2 * 2 + ch * 16),
                   W1h + (size_t)(h0 + row) * HIDDEN + ch * 8);
    }
    wait_async0();
    __syncthreads();

    v8f zero = {0.f, 0.f, 0.f, 0.f, 0.f, 0.f, 0.f, 0.f};
    v8f acc[4] = {zero, zero, zero, zero};

    v16h a = load_a_frag(X1, m0, 0, HIDDEN, lane);
    for (int kk = 0; kk < HIDDEN; kk += 32) {
        v16h an = a;
        if (kk + 32 < HIDDEN)
            an = load_a_frag(X1, m0, kk + 32, HIDDEN, lane);
#pragma unroll
        for (int j = 0; j < 4; ++j) {
            v16h bf = lds_b_frag(lW1, 16 * j, kk, LD2, lane);
            acc[j] = __builtin_amdgcn_wmma_f32_16x16x32_f16(
                false, a, false, bf, (short)0, acc[j], false, false);
        }
        a = an;
    }

    int n  = lane & 15;
    int mo = (lane & 16) ? 8 : 0;
#pragma unroll
    for (int j = 0; j < 4; ++j) {
        int col = h0 + 16 * j + n;
        float bb = b1[col];
#pragma unroll
        for (int v = 0; v < 8; ++v) {
            int row = m0 + mo + v;
            X2[(size_t)row * H2 + col] = (_Float16)sigmoidf_(acc[j][v] + bb);
        }
    }
}

// ---------------------------------------------------------------------------
// Kernel 4: layer 3 GEMV — out[b] = sigmoid(X2[b,:] . W2 + b2)
// ---------------------------------------------------------------------------
__global__ void final_kernel(const _Float16* __restrict__ X2,
                             const float* __restrict__ W2,
                             const float* __restrict__ b2,
                             float* __restrict__ out) {
    int b = blockIdx.x * blockDim.x + threadIdx.x;
    if (b >= BATCH) return;
    const _Float16* row = X2 + (size_t)b * H2;
    float s = b2[0];
#pragma unroll 8
    for (int h = 0; h < H2; ++h) s += (float)row[h] * W2[h];
    out[b] = sigmoidf_(s);
}

// ---------------------------------------------------------------------------
extern "C" void kernel_launch(void* const* d_in, const int* in_sizes, int n_in,
                              void* d_out, int out_size, void* d_ws, size_t ws_size,
                              hipStream_t stream) {
    (void)in_sizes; (void)n_in; (void)out_size; (void)ws_size;

    const int*   uid       = (const int*)d_in[0];
    const int*   iid       = (const int*)d_in[1];
    const float* item_know = (const float*)d_in[2];
    const float* priori    = (const float*)d_in[3];
    const float* condi_p   = (const float*)d_in[4];
    const float* condi_n   = (const float*)d_in[5];
    const float* item_diff = (const float*)d_in[6];
    const float* item_disc = (const float*)d_in[7];
    const float* W_user    = (const float*)d_in[8];
    const float* b_user    = (const float*)d_in[9];
    const float* W_item    = (const float*)d_in[10];
    const float* b_item    = (const float*)d_in[11];
    const float* W1        = (const float*)d_in[12];
    const float* b1        = (const float*)d_in[13];
    const float* W2        = (const float*)d_in[14];
    const float* b2        = (const float*)d_in[15];
    float* out = (float*)d_out;

    // Workspace layout (all chunks power-of-two sized -> 16B aligned for b128)
    _Float16* Xu   = (_Float16*)d_ws;                 // 8192*128
    _Float16* Xi   = Xu   + (size_t)BATCH * NKNOW;    // 8192*128
    _Float16* X1   = Xi   + (size_t)BATCH * NKNOW;    // 8192*512
    _Float16* X2   = X1   + (size_t)BATCH * HIDDEN;   // 8192*256
    _Float16* Wu16 = X2   + (size_t)BATCH * H2;       // 512*128
    _Float16* Wi16 = Wu16 + (size_t)HIDDEN * NKNOW;   // 512*128
    _Float16* W1h  = Wi16 + (size_t)HIDDEN * NKNOW;   // 256*512
    float*    disc = (float*)(W1h + (size_t)H2 * HIDDEN); // 8192

    // Weight conversion f32 -> f16
    convert_f16_kernel<<<(HIDDEN * NKNOW) / 256, 256, 0, stream>>>(W_user, Wu16, HIDDEN * NKNOW);
    convert_f16_kernel<<<(HIDDEN * NKNOW) / 256, 256, 0, stream>>>(W_item, Wi16, HIDDEN * NKNOW);
    convert_f16_kernel<<<(H2 * HIDDEN) / 256, 256, 0, stream>>>(W1, W1h, H2 * HIDDEN);

    // Gather + sigmoid + DAG scan + mask
    prep_kernel<<<BATCH / 256, 256, 0, stream>>>(uid, iid, item_know, priori,
                                                 condi_p, condi_n, item_diff,
                                                 item_disc, Xu, Xi, disc);

    // Layer 1: 128 M-blocks * 8 N-blocks
    gemm1_kernel<<<128 * 8, 128, 0, stream>>>(Xu, Xi, Wu16, Wi16,
                                              b_user, b_item, disc, X1);

    // Layer 2: 128 M-blocks * 4 N-blocks
    gemm2_kernel<<<128 * 4, 128, 0, stream>>>(X1, W1h, b1, X2);

    // Layer 3 GEMV
    final_kernel<<<BATCH / 256, 256, 0, stream>>>(X2, W2, b2, out);
}
